// MultiheadAttention_87290915323982
// MI455X (gfx1250) — matrix operand order repaired and checked by reading, not compiled, not run on hardware
//
#include <hip/hip_runtime.h>

// Problem constants (match reference)
constexpr int Bc = 4, Nc = 2048, Mc = 2048, Dc = 1024, Hc = 16, Pc = 64;
constexpr float INV_TEMP = 0.125f; // 1/sqrt(64)

typedef __bf16 bf16_t;
typedef __attribute__((ext_vector_type(16))) __bf16 v16bf;
typedef __attribute__((ext_vector_type(8)))  __bf16 v8bf;
typedef __attribute__((ext_vector_type(4)))  __bf16 v4bf;
typedef __attribute__((ext_vector_type(8)))  float  v8f;
typedef __attribute__((ext_vector_type(4)))  int    v4i;
typedef __attribute__((ext_vector_type(8)))  int    v8i;
typedef __attribute__((ext_vector_type(4)))  unsigned int v4u;

// ---------------------------------------------------------------------------
// Async global->LDS staging (GLOBAL_LOAD_ASYNC_TO_LDS_B128, ASYNCcnt)
// ---------------------------------------------------------------------------
#if __has_builtin(__builtin_amdgcn_global_load_async_to_lds_b128) && \
    __has_builtin(__builtin_amdgcn_s_wait_asynccnt)
#define USE_ASYNC_LDS 1
#else
#define USE_ASYNC_LDS 0
#endif

// Tensor Data Mover (TENSOR_LOAD_TO_LDS, TENSORcnt) — 6-arg clang-23 form
#if __has_builtin(__builtin_amdgcn_tensor_load_to_lds) && \
    __has_builtin(__builtin_amdgcn_s_wait_tensorcnt)
#define USE_TDM 1
#else
#define USE_TDM 0
#endif

__device__ __forceinline__ void copy16_to_lds(bf16_t* dst, const bf16_t* src) {
#if USE_ASYNC_LDS
  __builtin_amdgcn_global_load_async_to_lds_b128(
      (__attribute__((address_space(1))) v4i*)src,
      (__attribute__((address_space(3))) v4i*)dst, 0, 0);
#else
  *(v8bf*)dst = *(const v8bf*)src;
#endif
}

__device__ __forceinline__ void lds_stage_fence() {
#if USE_ASYNC_LDS
  __builtin_amdgcn_s_wait_asynccnt(0);
#endif
  __syncthreads();
}

#if USE_TDM
// Issue one TDM 2D tile load: tile_h rows x tile_w bf16 elements, global row
// stride row_stride_elems, packed contiguously at LDS byte address lds_addr.
// D# per CDNA5 ISA 8.3/8.4: group0 = {count=1, lds_addr, global_addr, type=2},
// group1 = {data_size=2B, tensor_dim0/1 = tile dims, tile_dim0/1, dim0 stride},
// groups 2/3 (and trailing group) zero: 2D tile, no iterate/gather/multicast.
__device__ __forceinline__ void tdm_load_2d(unsigned lds_addr, const void* gaddr,
                                            unsigned tile_w, unsigned tile_h,
                                            unsigned row_stride_elems) {
  const unsigned long long ga = (unsigned long long)gaddr;
  v4u g0;
  g0[0] = 1u;                                    // count=1 (valid user D#)
  g0[1] = lds_addr;                              // LDS byte address
  g0[2] = (unsigned)(ga & 0xFFFFFFFFu);          // global_addr[31:0]
  g0[3] = (unsigned)((ga >> 32) & 0x01FFFFFFu)   // global_addr[56:32]
        | (2u << 30);                            // type=2 ("image")
  v8i g1;
  g1[0] = (int)(1u << 16);                       // data_size=1 -> 2 bytes
  g1[1] = (int)(tile_w << 16);                   // tensor_dim0[15:0]
  g1[2] = (int)(tile_h << 16);                   // tensor_dim1[15:0]
  g1[3] = (int)(tile_w << 16);                   // tile_dim0
  g1[4] = (int)tile_h;                           // tile_dim1
  g1[5] = (int)row_stride_elems;                 // tensor_dim0_stride[31:0]
  g1[6] = 0;
  g1[7] = 0;
  const v4i z4 = {0, 0, 0, 0};
  const v8i z8 = {0, 0, 0, 0, 0, 0, 0, 0};
  __builtin_amdgcn_tensor_load_to_lds(g0, g1, z4, z4, z8, 0);
}
#endif

// ---------------------------------------------------------------------------
// DPP16 butterfly reductions over 16-lane rows (co-executable VALU, no LDS)
// ---------------------------------------------------------------------------
template <int CTRL>
__device__ __forceinline__ float dpp_movf(float x) {
  int r = __builtin_amdgcn_update_dpp(0, __builtin_bit_cast(int, x), CTRL, 0xF,
                                      0xF, true);
  return __builtin_bit_cast(float, r);
}

__device__ __forceinline__ float row16_max(float x) {
  x = fmaxf(x, dpp_movf<0xB1>(x));   // quad_perm(1,0,3,2): xor 1
  x = fmaxf(x, dpp_movf<0x4E>(x));   // quad_perm(2,3,0,1): xor 2
  x = fmaxf(x, dpp_movf<0x141>(x));  // row_half_mirror
  x = fmaxf(x, dpp_movf<0x140>(x));  // row_mirror
  return x;
}

__device__ __forceinline__ float row16_sum(float x) {
  x += dpp_movf<0xB1>(x);
  x += dpp_movf<0x4E>(x);
  x += dpp_movf<0x141>(x);
  x += dpp_movf<0x140>(x);
  return x;
}

// ---------------------------------------------------------------------------
// WMMA helpers (CDNA5 wave32, V_WMMA_F32_16X16X32_BF16)
// ---------------------------------------------------------------------------
__device__ __forceinline__ v8f wmma_bf16(v16bf a, v16bf b, v8f c) {
  return __builtin_amdgcn_wmma_f32_16x16x32_bf16(false, a, false, b, (short)0, c,
                                                 false, false);
}

// A-matrix 16x32 fragment (row-major source, K fast axis).
__device__ __forceinline__ v16bf load_A_frag(const bf16_t* base, int ld, int k0,
                                             int lane) {
  const int r  = lane & 15;
  const int kk = k0 + ((lane >> 4) << 3);
  const bf16_t* p = base + (size_t)r * ld + kk;
  v8bf lo = *(const v8bf*)(p);
  v8bf hi = *(const v8bf*)(p + 16);
  return __builtin_shufflevector(lo, hi, 0, 1, 2, 3, 4, 5, 6, 7,
                                 8, 9, 10, 11, 12, 13, 14, 15);
}

// B-matrix 32x16 fragment where B[k][n] = S[col0+n][k0+k] (S row-major).
__device__ __forceinline__ v16bf load_B_frag(const bf16_t* S, int ld, int col0,
                                             int k0, int lane) {
  const int c  = col0 + (lane & 15);
  const int kk = k0 + ((lane >> 4) << 3);
  const bf16_t* p = S + (size_t)c * ld + kk;
  v8bf lo = *(const v8bf*)(p);
  v8bf hi = *(const v8bf*)(p + 16);
  return __builtin_shufflevector(lo, hi, 0, 1, 2, 3, 4, 5, 6, 7,
                                 8, 9, 10, 11, 12, 13, 14, 15);
}

// ---------------------------------------------------------------------------
// Prep: mask (per-row) + fp32 -> bf16 convert, 4 elements/thread
// ---------------------------------------------------------------------------
__global__ void mask_cvt_kernel(const float* __restrict__ x,
                                const int* __restrict__ mask,
                                bf16_t* __restrict__ y, int cols, size_t total) {
  size_t i4 = ((size_t)blockIdx.x * blockDim.x + threadIdx.x) * 4;
  if (i4 >= total) return;
  const float4 x4 = *(const float4*)(x + i4);
  const float m = (float)mask[i4 / cols];
  v4bf y4 = {(bf16_t)(x4.x * m), (bf16_t)(x4.y * m), (bf16_t)(x4.z * m),
             (bf16_t)(x4.w * m)};
  *(v4bf*)(y + i4) = y4;
}

__global__ void cvt_kernel(const float* __restrict__ x, bf16_t* __restrict__ y,
                           size_t total) {
  size_t i4 = ((size_t)blockIdx.x * blockDim.x + threadIdx.x) * 4;
  if (i4 >= total) return;
  const float4 x4 = *(const float4*)(x + i4);
  v4bf y4 = {(bf16_t)x4.x, (bf16_t)x4.y, (bf16_t)x4.z, (bf16_t)x4.w};
  *(v4bf*)(y + i4) = y4;
}

// ---------------------------------------------------------------------------
// Projection GEMM: out = rowmask .* (X @ W^T + bias)
// Block = 4 waves sharing one 64-col W panel staged through LDS via the
// async-to-LDS path. OUT_MODE 0: bf16; 1: bf16 per-head-transposed; 2: f32.
// ---------------------------------------------------------------------------
template <int OUT_MODE>
__global__ __launch_bounds__(128) void gemm_proj(const bf16_t* __restrict__ X,
                                                 const bf16_t* __restrict__ W,
                                                 const float* __restrict__ bias,
                                                 const int* __restrict__ rowmask,
                                                 void* __restrict__ out) {
  __shared__ alignas(16) bf16_t ldsW[64][64];  // [out-col][k-within-block]
  const int tid = threadIdx.x;
  const int lane = tid & 31;
  const int wave = tid >> 5;
  const int row0 = blockIdx.y * 64 + wave * 16;
  const int col0 = blockIdx.x * 64;
  const bf16_t* A = X + (size_t)row0 * Dc;

  v8f acc[4] = {};
  for (int kb = 0; kb < Dc; kb += 64) {
    __syncthreads();  // previous tile fully consumed
#pragma unroll
    for (int i = 0; i < 4; ++i) {
      const int chunk = tid + i * 128;
      const int wr = chunk >> 3;            // 0..63 (output col)
      const int wo = (chunk & 7) * 8;       // 0..56 (k elements)
      copy16_to_lds(&ldsW[wr][wo], W + (size_t)(col0 + wr) * Dc + kb + wo);
    }
    lds_stage_fence();

#pragma unroll
    for (int ks = 0; ks < 64; ks += 32) {
      if (kb + ks + 32 < Dc)
        __builtin_prefetch(A + (size_t)(lane & 15) * Dc + kb + ks + 32, 0, 1);
      v16bf a = load_A_frag(A, Dc, kb + ks, lane);
#pragma unroll
      for (int t = 0; t < 4; ++t) {
        v16bf b = load_B_frag(&ldsW[0][0], 64, t * 16, ks, lane);
        acc[t] = wmma_bf16(a, b, acc[t]);
      }
    }
  }

  const int r_in = (lane >> 4) << 3;
  const int c_in = lane & 15;
#pragma unroll
  for (int t = 0; t < 4; ++t) {
#pragma unroll
    for (int j = 0; j < 8; ++j) {
      const int r = row0 + r_in + j;
      const int c = col0 + t * 16 + c_in;
      const float mval = (float)rowmask[r];
      const float v = (acc[t][j] + bias[c]) * mval;
      if constexpr (OUT_MODE == 0) {
        ((bf16_t*)out)[(size_t)r * Dc + c] = (bf16_t)v;
      } else if constexpr (OUT_MODE == 1) {
        const int bb = r / Mc, m = r % Mc;
        const int h = c / Pc, p = c % Pc;
        ((bf16_t*)out)[(((size_t)(bb * Hc + h)) * Pc + p) * Mc + m] = (bf16_t)v;
      } else {
        ((float*)out)[(size_t)r * Dc + c] = v;
      }
    }
  }
}

// ---------------------------------------------------------------------------
// Flash attention: block = 4 waves x 16 query rows of one (b,h).
// K tile (32x64) and V tile (64x32, pre-transposed) staged in LDS by the
// Tensor Data Mover (wave 0 issues two D#s, TENSORcnt) with async/ds fallback.
// Per 32-key step: 4 WMMAs for S = Q@K^T, DPP online softmax, P-tile relayout
// C->A through LDS, 4 WMMAs for O += P@V.
// ---------------------------------------------------------------------------
__global__ __launch_bounds__(128) void flash_attn(const bf16_t* __restrict__ Q,
                                                  const bf16_t* __restrict__ Kp,
                                                  const bf16_t* __restrict__ Vt,
                                                  const int* __restrict__ mask_key,
                                                  bf16_t* __restrict__ Oo) {
  __shared__ alignas(16) bf16_t ldsK[32][64];      // [key][head-dim]
  __shared__ alignas(16) bf16_t ldsV[64][32];      // [head-dim][key]
  __shared__ alignas(16) bf16_t lds_p[4][16][32];  // per-wave P tile
  const int tid = threadIdx.x;
  const int lane = tid & 31;
  const int wave = tid >> 5;
  const int b = blockIdx.z, h = blockIdx.y;
  const int row0 = blockIdx.x * 64 + wave * 16;

  const bf16_t* Qb = Q + ((size_t)(b * Nc + row0)) * Dc + h * Pc;  // 16x64, ld=Dc
  const bf16_t* Kb = Kp + (size_t)b * Mc * Dc + h * Pc;            // Mx64,  ld=Dc
  const bf16_t* Vb = Vt + ((size_t)(b * Hc + h)) * Pc * Mc;        // 64xM,  ld=Mc

  const v16bf qa0 = load_A_frag(Qb, Dc, 0, lane);
  const v16bf qa1 = load_A_frag(Qb, Dc, 32, lane);

  float m_run[8], l_run[8];
  v8f acc[4] = {};
#pragma unroll
  for (int j = 0; j < 8; ++j) { m_run[j] = -1e30f; l_run[j] = 0.f; }

  const int c_in = lane & 15;
  const int r_in = (lane >> 4) << 3;

#if USE_TDM
  const unsigned ldsK_addr =
      (unsigned)(unsigned long long)(__attribute__((address_space(3))) void*)&ldsK[0][0];
  const unsigned ldsV_addr =
      (unsigned)(unsigned long long)(__attribute__((address_space(3))) void*)&ldsV[0][0];
#endif

  for (int m0 = 0; m0 < Mc; m0 += 32) {
    __syncthreads();  // previous K/V tiles fully consumed
#if USE_TDM
    if (wave == 0) {
      tdm_load_2d(ldsK_addr, Kb + (size_t)m0 * Dc, /*w*/ 64, /*h*/ 32, Dc);
      tdm_load_2d(ldsV_addr, Vb + m0,              /*w*/ 32, /*h*/ 64, Mc);
      __builtin_amdgcn_s_wait_tensorcnt(0);
    }
    __syncthreads();
#else
#pragma unroll
    for (int i = 0; i < 2; ++i) {
      const int chunk = tid + i * 128;
      const int kr = chunk >> 3;           // key row 0..31
      const int ko = (chunk & 7) * 8;      // head-dim elements
      copy16_to_lds(&ldsK[kr][ko], Kb + (size_t)(m0 + kr) * Dc + ko);
    }
#pragma unroll
    for (int i = 0; i < 2; ++i) {
      const int chunk = tid + i * 128;
      const int vr = chunk >> 2;           // head-dim row 0..63
      const int vo = (chunk & 3) * 8;      // key elements
      copy16_to_lds(&ldsV[vr][vo], Vb + (size_t)vr * Mc + m0 + vo);
    }
    lds_stage_fence();
#endif

    v8f s0 = {}, s1 = {};
    s0 = wmma_bf16(qa0, load_B_frag(&ldsK[0][0], 64, 0, 0, lane), s0);
    s0 = wmma_bf16(qa1, load_B_frag(&ldsK[0][0], 64, 0, 32, lane), s0);
    s1 = wmma_bf16(qa0, load_B_frag(&ldsK[0][0], 64, 16, 0, lane), s1);
    s1 = wmma_bf16(qa1, load_B_frag(&ldsK[0][0], 64, 16, 32, lane), s1);

    const float mk0 = mask_key[b * Mc + m0 + c_in] ? 0.f : -1e30f;
    const float mk1 = mask_key[b * Mc + m0 + 16 + c_in] ? 0.f : -1e30f;

#pragma unroll
    for (int j = 0; j < 8; ++j) {
      const float x0 = s0[j] * INV_TEMP + mk0;
      const float x1 = s1[j] * INV_TEMP + mk1;
      const float mx = row16_max(fmaxf(x0, x1));
      const float mnew = fmaxf(m_run[j], mx);
      const float alpha = __expf(m_run[j] - mnew);
      m_run[j] = mnew;
      const float p0 = __expf(x0 - mnew);
      const float p1 = __expf(x1 - mnew);
      const float ps = row16_sum(p0 + p1);
      l_run[j] = l_run[j] * alpha + ps;
#pragma unroll
      for (int t = 0; t < 4; ++t) acc[t][j] *= alpha;
      lds_p[wave][r_in + j][c_in] = (bf16_t)p0;
      lds_p[wave][r_in + j][c_in + 16] = (bf16_t)p1;
    }

    const v16bf pa = load_A_frag(&lds_p[wave][0][0], 32, 0, lane);
#pragma unroll
    for (int t = 0; t < 4; ++t)
      acc[t] = wmma_bf16(pa, load_B_frag(&ldsV[0][0], 32, t * 16, 0, lane),
                         acc[t]);
  }

#pragma unroll
  for (int t = 0; t < 4; ++t)
#pragma unroll
    for (int j = 0; j < 8; ++j) {
      const float o = acc[t][j] / l_run[j];
      Oo[((size_t)(b * Nc + row0 + r_in + j)) * Dc + h * Pc + t * 16 + c_in] =
          (bf16_t)o;
    }
}

// ---------------------------------------------------------------------------
// Host-side launcher
// ---------------------------------------------------------------------------
extern "C" void kernel_launch(void* const* d_in, const int* in_sizes, int n_in,
                              void* d_out, int out_size, void* d_ws,
                              size_t ws_size, hipStream_t stream) {
  (void)in_sizes; (void)n_in; (void)out_size; (void)ws_size;
  const float* queries = (const float*)d_in[0];
  const float* keys    = (const float*)d_in[1];
  const float* values  = (const float*)d_in[2];
  const int*   mask_q  = (const int*)d_in[3];
  const int*   mask_k  = (const int*)d_in[4];
  const float* Wq = (const float*)d_in[5];
  const float* bq = (const float*)d_in[6];
  const float* Wk = (const float*)d_in[7];
  const float* bk = (const float*)d_in[8];
  const float* Wv = (const float*)d_in[9];
  const float* bv = (const float*)d_in[10];
  const float* Wo = (const float*)d_in[11];
  const float* bo = (const float*)d_in[12];

  bf16_t* ws = (bf16_t*)d_ws;
  const size_t SZX = (size_t)Bc * Nc * Dc;  // 8M elements
  const size_t DD  = (size_t)Dc * Dc;       // 1M elements
  bf16_t* Xq = ws;
  bf16_t* Xk = ws + SZX;
  bf16_t* Xv = ws + 2 * SZX;
  bf16_t* Qp = ws + 3 * SZX;
  bf16_t* KpB = ws + 4 * SZX;
  bf16_t* VtB = ws + 5 * SZX;
  bf16_t* Wq16 = ws + 6 * SZX;
  bf16_t* Wk16 = Wq16 + DD;
  bf16_t* Wv16 = Wq16 + 2 * DD;
  bf16_t* Wo16 = Wq16 + 3 * DD;
  bf16_t* AO = Xq;  // reuse: Xq dead after Q projection

  const dim3 cb(256);
  const dim3 cg((unsigned)((SZX / 4 + 255) / 256));
  mask_cvt_kernel<<<cg, cb, 0, stream>>>(queries, mask_q, Xq, Dc, SZX);
  mask_cvt_kernel<<<cg, cb, 0, stream>>>(keys,    mask_k, Xk, Dc, SZX);
  mask_cvt_kernel<<<cg, cb, 0, stream>>>(values,  mask_k, Xv, Dc, SZX);
  const dim3 wgrid((unsigned)((DD / 4 + 255) / 256));
  cvt_kernel<<<wgrid, cb, 0, stream>>>(Wq, Wq16, DD);
  cvt_kernel<<<wgrid, cb, 0, stream>>>(Wk, Wk16, DD);
  cvt_kernel<<<wgrid, cb, 0, stream>>>(Wv, Wv16, DD);
  cvt_kernel<<<wgrid, cb, 0, stream>>>(Wo, Wo16, DD);

  const dim3 gb(128);
  const dim3 gg(Dc / 64, (Bc * Nc) / 64);
  gemm_proj<0><<<gg, gb, 0, stream>>>(Xq, Wq16, bq, mask_q, Qp);
  gemm_proj<0><<<gg, gb, 0, stream>>>(Xk, Wk16, bk, mask_k, KpB);
  gemm_proj<1><<<gg, gb, 0, stream>>>(Xv, Wv16, bv, mask_k, VtB);

  const dim3 fg(Nc / 64, Hc, Bc);
  flash_attn<<<fg, gb, 0, stream>>>(Qp, KpB, VtB, mask_k, AO);

  gemm_proj<2><<<gg, gb, 0, stream>>>(AO, Wo16, bo, mask_q, d_out);
}
